// GraphConvWithSAT_52999896433219
// MI455X (gfx1250) — matrix operand order repaired and checked by reading, not compiled, not run on hardware
//
#include <hip/hip_runtime.h>
#include <hip/hip_bf16.h>
#include <math.h>

// ---------------------------------------------------------------------------
// GraphConv with spatial attention, refactored:
//   M[k,f,o'] = sum_o theta[k,f,o] * W[o*K+k, o']            (tiny)
//   G[b, p=(k*512+m), j=(t*64+o')] = sum_f x[b,m,f,t]*M[k,f,o']   (9.7 GF)
//   y[b,n,o',t] = tanh( sum_p At[b,n,p]*G[b,p,j] + bias[o'] )     (77 GF)
// All big contractions run on v_wmma_f32_16x16x32_f16 (wave32).
// ---------------------------------------------------------------------------

#define Bsz 32
#define Nn  512
#define Ff  64
#define Tt  24
#define Kk  3
#define Oo  64
#define Pdim (Kk * Nn)      // 1536 contraction dim of main GEMM
#define Jdim (Tt * Oo)      // 1536 column dim of main GEMM

typedef __attribute__((ext_vector_type(16))) _Float16 v16h;
typedef __attribute__((ext_vector_type(8)))  _Float16 v8h;
typedef __attribute__((ext_vector_type(8)))  float    v8f;

__device__ __forceinline__ v8f zero8() {
    v8f z;
#pragma unroll
    for (int i = 0; i < 8; ++i) z[i] = 0.0f;
    return z;
}

__device__ __forceinline__ v16h mk16(v8h lo, v8h hi) {
    v16h r;
#pragma unroll
    for (int i = 0; i < 8; ++i) { r[i] = lo[i]; r[i + 8] = hi[i]; }
    return r;
}

__device__ __forceinline__ v8f wmma_f16(v16h a, v16h b, v8f c) {
    // D = A(16x32 f16) * B(32x16 f16) + C(16x16 f32)
    return __builtin_amdgcn_wmma_f32_16x16x32_f16(
        /*neg_a=*/false, a, /*neg_b=*/false, b,
        /*c_mod=*/(short)0, c, /*reuse_a=*/false, /*reuse_b=*/false);
}

// ---------------------------------------------------------------------------
// Kernel 1: fold theta and W:  Mt[c=(k*64+o')][f] = sum_o theta[k,f,o]*W[o*3+k,o']
// Also writes kl = 0.0 (second tuple output).
// grid: 192 blocks (k*64+f), block: 64 threads (o')
// ---------------------------------------------------------------------------
__global__ void fold_theta_w_kernel(const float* __restrict__ theta,
                                    const float* __restrict__ W,
                                    _Float16* __restrict__ Mt,
                                    float* __restrict__ kl_out) {
    int kf = blockIdx.x;
    int k = kf >> 6, f = kf & 63;
    int op = threadIdx.x; // o'
    float acc = 0.0f;
    const float* th = theta + ((size_t)(k * Ff + f)) * Oo;
#pragma unroll 4
    for (int o = 0; o < Oo; ++o) {
        acc += th[o] * W[(size_t)(o * Kk + k) * Oo + op];
    }
    Mt[(size_t)(k * Oo + op) * Ff + f] = (_Float16)acc;
    if (kf == 0 && op == 0) kl_out[0] = 0.0f;
}

// ---------------------------------------------------------------------------
// Kernel 2: row softmax over sa[b,k,m,:] and transposed f16 store:
//   At[b][n][k*512+m] = softmax(sa[b,k,m,:])[n]
// grid: (32 m-groups, 3 k, 32 b), block: 512 (16 waves; one wave per row m)
// ---------------------------------------------------------------------------
__global__ void softmax_t_kernel(const float* __restrict__ sa,
                                 _Float16* __restrict__ At) {
    __shared__ _Float16 lds[16 * 520];

    int mg  = blockIdx.x;          // 16-row group
    int k   = blockIdx.y;
    int b   = blockIdx.z;
    int w   = threadIdx.x >> 5;    // wave id = row within group
    int lane = threadIdx.x & 31;

    int m = mg * 16 + w;
    const float* row = sa + (((size_t)(b * Kk + k) * Nn + m) * Nn);

    float v[16];
    float mx = -INFINITY;
#pragma unroll
    for (int i = 0; i < 16; ++i) {
        v[i] = row[lane * 16 + i];
        mx = fmaxf(mx, v[i]);
    }
#pragma unroll
    for (int off = 16; off > 0; off >>= 1)
        mx = fmaxf(mx, __shfl_xor(mx, off, 32));

    float s = 0.0f;
#pragma unroll
    for (int i = 0; i < 16; ++i) { v[i] = __expf(v[i] - mx); s += v[i]; }
#pragma unroll
    for (int off = 16; off > 0; off >>= 1)
        s += __shfl_xor(s, off, 32);
    float inv = 1.0f / s;

#pragma unroll
    for (int i = 0; i < 16; ++i)
        lds[w * 520 + lane * 16 + i] = (_Float16)(v[i] * inv);

    __syncthreads();

    // Each thread writes one column n: 16 contiguous halves (m direction).
    int n = threadIdx.x;
    v16h out;
#pragma unroll
    for (int i = 0; i < 16; ++i) out[i] = lds[i * 520 + n];
    *(v16h*)(At + ((size_t)(b * Nn + n)) * Pdim + k * Nn + mg * 16) = out;
}

// ---------------------------------------------------------------------------
// Kernel 2b: xt[b][t][m][f] = (f16) x[b][m][f][t]   (LDS transpose per (b,m))
// grid: (512 m, 32 b), block: 256
// ---------------------------------------------------------------------------
__global__ void x_transpose_kernel(const float* __restrict__ x,
                                   _Float16* __restrict__ xt) {
    __shared__ float lds[Ff * Tt]; // 1536
    int m = blockIdx.x, b = blockIdx.y;
    const float* src = x + ((size_t)(b * Nn + m)) * (Ff * Tt);
#pragma unroll
    for (int i = 0; i < 6; ++i) {
        int j = threadIdx.x + i * 256;
        lds[j] = src[j];
    }
    __syncthreads();
#pragma unroll
    for (int i = 0; i < 6; ++i) {
        int j = threadIdx.x + i * 256;
        int t = j >> 6, f = j & 63;
        xt[((size_t)(b * Tt + t) * Nn + m) * Ff + f] = (_Float16)lds[f * Tt + t];
    }
}

// ---------------------------------------------------------------------------
// Kernel 3: G-prep GEMM (per (b,t)):  D[c,m] = sum_f Mt[c,f] * xt[b,t][m,f]
// Stores Gt[b][j=t*64+o'][p=k*512+m] (contiguous in m across lanes).
// grid: (12 c-tiles, 24 t, 32 b), block: 256 = 8 waves; wave w -> m in [w*64, w*64+64)
// ---------------------------------------------------------------------------
__global__ void g_prep_kernel(const _Float16* __restrict__ xt,
                              const _Float16* __restrict__ Mt,
                              _Float16* __restrict__ Gt) {
    int lane = threadIdx.x & 31;
    int w    = threadIdx.x >> 5;
    int cb   = blockIdx.x * 16;
    int t    = blockIdx.y;
    int b    = blockIdx.z;

    const _Float16* X = xt + ((size_t)(b * Tt + t) * Nn) * Ff;

    int l  = lane & 15;
    int hi = lane >> 4;

    v8f acc[4];
#pragma unroll
    for (int i = 0; i < 4; ++i) acc[i] = zero8();

#pragma unroll
    for (int f0 = 0; f0 < Ff; f0 += 32) {
        // A fragment: Mt rows c=cb+l, K=f contiguous (lanes<16: K 0-15; lanes>=16: 16-31)
        const _Float16* ap = Mt + (size_t)(cb + l) * Ff + f0 + hi * 16;
        v16h a = *(const v16h*)ap;
#pragma unroll
        for (int i = 0; i < 4; ++i) {
            int m = w * 64 + i * 16 + l;
            // B fragment: col m, K=f contiguous (lanes<16: K 0-15; lanes>=16: 16-31)
            v16h bf = *(const v16h*)(X + (size_t)m * Ff + f0 + hi * 16);
            acc[i] = wmma_f16(a, bf, acc[i]);
        }
    }

#pragma unroll
    for (int i = 0; i < 4; ++i) {
        int m = w * 64 + i * 16 + (lane & 15);
#pragma unroll
        for (int r = 0; r < 8; ++r) {
            int c = cb + r + hi * 8;            // c = k*64 + o'
            int o = c & 63, kk = c >> 6;
            Gt[((size_t)b * Jdim + t * Oo + o) * Pdim + kk * Nn + m] =
                (_Float16)acc[i][r];
        }
    }
}

// ---------------------------------------------------------------------------
// Kernel 4: main GEMM per b:  C[n,j] = sum_p At[n,p]*Gt[j,p]; epilogue
//   y[b,n,o',t] = tanh(C[n, t*64+o'] + bias[o'])
// grid: (6 j-blocks, 8 n-blocks, 32 b), block 256 = 8 waves in 2(n) x 4(j);
// each wave: 32n x 64j = 2x4 WMMA tiles, K-loop 48 steps of 32.
// ---------------------------------------------------------------------------
__global__ void main_gemm_kernel(const _Float16* __restrict__ At,
                                 const _Float16* __restrict__ Gt,
                                 const float* __restrict__ bias,
                                 float* __restrict__ y) {
    int lane = threadIdx.x & 31;
    int w    = threadIdx.x >> 5;
    int wn   = w >> 2;             // 0..1
    int wj   = w & 3;              // 0..3
    int b    = blockIdx.z;
    int n0   = blockIdx.y * 64 + wn * 32;
    int j0   = blockIdx.x * 256 + wj * 64;

    const _Float16* Ab = At + (size_t)b * Nn * Pdim;
    const _Float16* Gb = Gt + (size_t)b * Jdim * Pdim;

    int l  = lane & 15;
    int hi = lane >> 4;

    v8f acc[2][4];
#pragma unroll
    for (int i = 0; i < 2; ++i)
#pragma unroll
        for (int j = 0; j < 4; ++j) acc[i][j] = zero8();

#pragma unroll 2
    for (int p0 = 0; p0 < Pdim; p0 += 32) {
        if (p0 + 64 < Pdim) {
            // L2 prefetch of next-next K-slab (global_prefetch_b8)
            __builtin_prefetch(Ab + (size_t)(n0 + l) * Pdim + p0 + 64, 0, 1);
            __builtin_prefetch(Gb + (size_t)(j0 + l) * Pdim + p0 + 64, 0, 1);
        }
        v16h a[2];
#pragma unroll
        for (int i = 0; i < 2; ++i) {
            const _Float16* ap = Ab + (size_t)(n0 + i * 16 + l) * Pdim + p0 + hi * 16;
            a[i] = *(const v16h*)ap;
        }
#pragma unroll
        for (int jt = 0; jt < 4; ++jt) {
            v16h bf = *(const v16h*)(Gb + (size_t)(j0 + jt * 16 + l) * Pdim +
                                     p0 + hi * 16);
#pragma unroll
            for (int i = 0; i < 2; ++i)
                acc[i][jt] = wmma_f16(a[i], bf, acc[i][jt]);
        }
    }

    // Epilogue: bias + tanh + transpose to [b,n,o,t]
#pragma unroll
    for (int i = 0; i < 2; ++i) {
#pragma unroll
        for (int jt = 0; jt < 4; ++jt) {
            int j = j0 + jt * 16 + (lane & 15);
            int t = j >> 6, o = j & 63;
            float bb = bias[o];
#pragma unroll
            for (int r = 0; r < 8; ++r) {
                int n = n0 + i * 16 + r + hi * 8;
                float v = tanhf(acc[i][jt][r] + bb);
                y[(((size_t)b * Nn + n) * Oo + o) * Tt + t] = v;
            }
        }
    }
}

// ---------------------------------------------------------------------------
extern "C" void kernel_launch(void* const* d_in, const int* in_sizes, int n_in,
                              void* d_out, int out_size, void* d_ws, size_t ws_size,
                              hipStream_t stream) {
    const float* x     = (const float*)d_in[0]; // [32,512,64,24]
    const float* sa    = (const float*)d_in[1]; // [32,3,512,512]
    const float* theta = (const float*)d_in[2]; // [3,64,64]
    const float* W     = (const float*)d_in[3]; // [192,64]
    const float* bias  = (const float*)d_in[4]; // [64]

    float* y   = (float*)d_out;                       // [32,512,64,24]
    float* klp = y + (size_t)Bsz * Nn * Oo * Tt;      // kl scalar

    // Workspace layout (all 256B-aligned offsets)
    char* ws = (char*)d_ws;
    const size_t at_bytes = (size_t)Bsz * Nn * Pdim * sizeof(_Float16);   // 50.3 MB
    const size_t gt_bytes = (size_t)Bsz * Jdim * Pdim * sizeof(_Float16); // 151.0 MB
    const size_t xt_bytes = (size_t)Bsz * Tt * Nn * Ff * sizeof(_Float16);// 50.3 MB
    _Float16* At = (_Float16*)ws;
    _Float16* Gt = (_Float16*)(ws + at_bytes);
    _Float16* xt = (_Float16*)(ws + at_bytes + gt_bytes);
    _Float16* Mt = (_Float16*)(ws + at_bytes + gt_bytes + xt_bytes);

    // 1) Fold theta*W into Mt; write kl=0
    fold_theta_w_kernel<<<dim3(Kk * Ff), dim3(Oo), 0, stream>>>(theta, W, Mt, klp);

    // 2) Softmax + transpose to f16 At[b][n][k*512+m]
    softmax_t_kernel<<<dim3(Nn / 16, Kk, Bsz), dim3(512), 0, stream>>>(sa, At);

    // 2b) x -> xt[b][t][m][f] f16
    x_transpose_kernel<<<dim3(Nn, Bsz), dim3(256), 0, stream>>>(x, xt);

    // 3) G = x * M   (WMMA), stored transposed Gt[b][j][p]
    g_prep_kernel<<<dim3(Jdim / Oo / 2, Tt, Bsz), dim3(256), 0, stream>>>(xt, Mt, Gt);

    // 4) Main GEMM + tanh epilogue (WMMA)
    main_gemm_kernel<<<dim3(Jdim / 256, Nn / 64, Bsz), dim3(256), 0, stream>>>(
        At, Gt, bias, y);
}
